// HungarianMatcher_4741643895265
// MI455X (gfx1250) — matrix operand order repaired and checked by reading, not compiled, not run on hardware
//
#include <hip/hip_runtime.h>
#include <hip/hip_bf16.h>
#include <math.h>
#include <stdint.h>

// Problem dims (fixed by the reference)
#define BS 16
#define Q 900
#define NC 92
#define NCP 96          // NC padded to multiple of 32 for WMMA K
#define T 256
#define SUMT (BS * T)   // 4096
#define NROWS (BS * Q)  // 14400

typedef __attribute__((ext_vector_type(16))) _Float16 v16h;
typedef __attribute__((ext_vector_type(8)))  float    v8f;

// ---------------------------------------------------------------------------
// Kernel 1: row softmax of logits -> f16 probs, K padded 92 -> 96 with zeros
// ---------------------------------------------------------------------------
__global__ __launch_bounds__(256) void hm_softmax_kernel(
    const float* __restrict__ logits,   // [NROWS][NC]
    _Float16* __restrict__ probs)       // [NROWS][NCP]
{
    int r = blockIdx.x * blockDim.x + threadIdx.x;
    if (r >= NROWS) return;
    const float* row = logits + (size_t)r * NC;
    float mx = -3.0e38f;
    for (int k = 0; k < NC; ++k) mx = fmaxf(mx, row[k]);
    float s = 0.0f;
    for (int k = 0; k < NC; ++k) s += __expf(row[k] - mx);
    float inv = 1.0f / s;
    _Float16* prow = probs + (size_t)r * NCP;
    for (int k = 0; k < NCP; ++k) {
        float p = (k < NC) ? (__expf(row[k] - mx) * inv) : 0.0f;
        prow[k] = (_Float16)p;
    }
}

// ---------------------------------------------------------------------------
// Kernel 2: fused cost matrix.
// Block = 256 threads (8 waves). blockIdx.y = row tile (16 queries),
// blockIdx.x = group of 8 col tiles (8 waves x 16 targets = 128 targets).
// Staging via GLOBAL_LOAD_ASYNC_TO_LDS_B128 (ASYNCcnt), class cost via
// 3x V_WMMA_F32_16X16X32_F16 (K = 96) with one-hot B synthesized from labels
// in registers, GIoU + L1 fused on the accumulator fragment.
// ---------------------------------------------------------------------------
__global__ __launch_bounds__(256) void hm_cost_kernel(
    const _Float16* __restrict__ probs,    // [NROWS][NCP] f16
    const float* __restrict__ pred_boxes,  // [NROWS][4] cxcywh
    const int* __restrict__ labels,        // [SUMT]
    const float* __restrict__ tboxes,      // [SUMT][4] cxcywh
    float* __restrict__ C)                 // [NROWS][SUMT]
{
    __shared__ _Float16 sA[16 * NCP];  // probs tile, 16 rows x 96 K (3072 B)
    __shared__ float    sPB[16 * 4];   // pred boxes for the 16 rows (256 B)

    const int tid     = threadIdx.x;
    const int rowBase = blockIdx.y * 16;

    // Prefetch the target boxes this block's 128 columns will consume
    // (emits global_prefetch_b8; speculative, no counter impact).
    __builtin_prefetch(tboxes + ((size_t)blockIdx.x * 128 + (tid >> 1)) * 4, 0, 0);

    // Async global -> LDS staging (16 B per enabled lane, ASYNCcnt-tracked).
    // Generic pointers to LDS carry the LDS byte offset in their low 32 bits.
    if (tid < 192) {
        unsigned lds = (unsigned)(uintptr_t)(&sA[0]) + (unsigned)tid * 16u;
        const void* g = (const char*)probs + (size_t)rowBase * NCP * 2 + (size_t)tid * 16u;
        asm volatile("global_load_async_to_lds_b128 %0, %1, off"
                     :: "v"(lds), "v"(g) : "memory");
    } else if (tid < 208) {
        const int t = tid - 192;
        unsigned lds = (unsigned)(uintptr_t)(&sPB[0]) + (unsigned)t * 16u;
        const void* g = (const char*)pred_boxes + (size_t)rowBase * 16 + (size_t)t * 16u;
        asm volatile("global_load_async_to_lds_b128 %0, %1, off"
                     :: "v"(lds), "v"(g) : "memory");
    }
    asm volatile("s_wait_asynccnt 0" ::: "memory");
    __syncthreads();

    const int wave    = tid >> 5;
    const int lane    = tid & 31;
    const int colTile = blockIdx.x * 8 + wave;   // 0..255
    const int colBase = colTile * 16;
    const int N       = lane & 15;               // C/D column owned by lane
    const int col     = colBase + N;
    const int hi      = lane >> 4;               // 0: lanes 0-15, 1: lanes 16-31
    const int M0      = lane & 15;               // A row owned by lane

    const int   lab = labels[col];
    const float tcx = tboxes[col * 4 + 0];
    const float tcy = tboxes[col * 4 + 1];
    const float tw  = tboxes[col * 4 + 2];
    const float th  = tboxes[col * 4 + 3];

    v8f acc = {};
    #pragma unroll
    for (int kt = 0; kt < 3; ++kt) {
        v16h a, b;
        const int kbaseA = kt * 32 + (hi ? 8 : 0);
        const int kbaseB = kt * 32 + (hi ? 8 : 0);
        #pragma unroll
        for (int h = 0; h < 16; ++h) {
            // 16-bit A 16x32 fragment: lanes 0-15 hold K = {0..7,16..23},
            // lanes 16-31 hold K = {8..15,24..31} (ISA 7.12.2).
            const int kA = kbaseA + ((h >> 3) << 4) + (h & 7);
            a[h] = sA[M0 * NCP + kA];
            // 16-bit B 32x16 fragment: same K order as A.
            const int kB = kbaseB + ((h >> 3) << 4) + (h & 7);
            b[h] = (kB == lab) ? (_Float16)1.0f : (_Float16)0.0f;
        }
        acc = __builtin_amdgcn_wmma_f32_16x16x32_f16(
            /*neg_a=*/false, a, /*neg_b=*/false, b,
            /*c_mod=*/(short)0, acc, /*reuse_a=*/false, /*reuse_b=*/false);
    }

    // Target box in xyxy
    const float tx1 = tcx - 0.5f * tw, ty1 = tcy - 0.5f * th;
    const float tx2 = tcx + 0.5f * tw, ty2 = tcy + 0.5f * th;
    const float a2  = (tx2 - tx1) * (ty2 - ty1);

    #pragma unroll
    for (int i = 0; i < 8; ++i) {
        const int M = i + (hi ? 8 : 0);      // C/D row: VGPR i -> M (or M+8)
        const float pcx = sPB[M * 4 + 0], pcy = sPB[M * 4 + 1];
        const float pw  = sPB[M * 4 + 2], ph  = sPB[M * 4 + 3];
        const float px1 = pcx - 0.5f * pw, py1 = pcy - 0.5f * ph;
        const float px2 = pcx + 0.5f * pw, py2 = pcy + 0.5f * ph;
        const float a1  = (px2 - px1) * (py2 - py1);
        const float ltx = fmaxf(px1, tx1), lty = fmaxf(py1, ty1);
        const float rbx = fminf(px2, tx2), rby = fminf(py2, ty2);
        const float iw  = fmaxf(rbx - ltx, 0.0f), ih = fmaxf(rby - lty, 0.0f);
        const float inter = iw * ih;
        const float uni   = a1 + a2 - inter;
        const float iou   = inter / uni;
        const float ex1 = fminf(px1, tx1), ey1 = fminf(py1, ty1);
        const float ex2 = fmaxf(px2, tx2), ey2 = fmaxf(py2, ty2);
        const float ea  = (ex2 - ex1) * (ey2 - ey1);
        const float giou = iou - (ea - uni) / ea;
        const float l1 = fabsf(pcx - tcx) + fabsf(pcy - tcy) +
                         fabsf(pw - tw) + fabsf(ph - th);
        const float cost = 5.0f * l1 - acc[i] - 2.0f * giou;
        C[(size_t)(rowBase + M) * SUMT + col] = cost;
    }
}

// ---------------------------------------------------------------------------
// Kernel 3: Jonker-Volgenant assignment, one workgroup per image.
// Solves on the transposed 256x900 slice (matches reference: n=256 <= m=900).
// Column scans + argmin parallelized over 256 threads; state in LDS.
// ---------------------------------------------------------------------------
__global__ __launch_bounds__(256) void hm_lsa_kernel(
    const float* __restrict__ C,     // [NROWS][SUMT]
    float* __restrict__ out_rows,    // [BS][T] (query indices)
    float* __restrict__ out_cols)    // [BS][T] (target indices)
{
    const int b   = blockIdx.x;
    const int tid = threadIdx.x;
    const int n = T;    // 256 rows (targets) after transpose
    const int m = Q;    // 900 cols (queries)
    const float INF = 1.0e30f;

    __shared__ float u[T + 1];
    __shared__ float v[Q + 1];
    __shared__ float minv[Q + 1];
    __shared__ int   p[Q + 1];
    __shared__ int   way[Q + 1];
    __shared__ unsigned char used[Q + 1];
    __shared__ float red_val[256];
    __shared__ int   red_idx[256];
    __shared__ int   s_j0;

    // Image slice base: cost.T[ti][q] = C[b*Q + q][b*T + ti]
    const float* Cb = C + (size_t)b * Q * SUMT + (size_t)b * T;

    for (int j = tid; j <= m; j += 256) { v[j] = 0.0f; p[j] = 0; }
    for (int i = tid; i <= n; i += 256) { u[i] = 0.0f; }
    __syncthreads();

    for (int i = 1; i <= n; ++i) {
        if (tid == 0) { p[0] = i; s_j0 = 0; }
        for (int j = tid; j <= m; j += 256) {
            minv[j] = INF; way[j] = 0; used[j] = 0;
        }
        __syncthreads();

        int done = 0;
        for (int iter = 0; iter < m + 2 && !done; ++iter) {
            const int j0 = s_j0;
            if (tid == 0) used[j0] = 1;
            __syncthreads();
            const int   i0  = p[j0];
            const float ui0 = u[i0];

            float bestv = INF;
            int   bestj = 0;
            for (int j = tid + 1; j <= m; j += 256) {
                if (!used[j]) {
                    const float cur =
                        Cb[(size_t)(j - 1) * SUMT + (i0 - 1)] - ui0 - v[j];
                    if (cur < minv[j]) { minv[j] = cur; way[j] = j0; }
                    if (minv[j] < bestv) { bestv = minv[j]; bestj = j; }
                }
            }
            red_val[tid] = bestv;
            red_idx[tid] = bestj;
            __syncthreads();
            for (int s = 128; s > 0; s >>= 1) {
                if (tid < s && red_val[tid + s] < red_val[tid]) {
                    red_val[tid] = red_val[tid + s];
                    red_idx[tid] = red_idx[tid + s];
                }
                __syncthreads();
            }
            const float delta = red_val[0];
            const int   j1    = red_idx[0];

            for (int j = tid; j <= m; j += 256) {
                if (used[j]) { u[p[j]] += delta; v[j] -= delta; }
                else         { minv[j] -= delta; }
            }
            __syncthreads();
            if (tid == 0) s_j0 = j1;
            if (p[j1] == 0) done = 1;
            __syncthreads();
        }

        // Augment along the alternating path (serial, thread 0)
        if (tid == 0) {
            int j0 = s_j0;
            while (j0 != 0) {
                const int jn = way[j0];
                p[j0] = p[jn];
                j0 = jn;
            }
        }
        __syncthreads();
    }

    // Emit assignments sorted by query index (j ascending). After the
    // reference's transpose-back: row = query (j-1), col = target (p[j]-1).
    if (tid == 0) {
        int cnt = 0;
        for (int j = 1; j <= m; ++j) {
            if (p[j] != 0) {
                out_rows[b * T + cnt] = (float)(j - 1);
                out_cols[b * T + cnt] = (float)(p[j] - 1);
                ++cnt;
            }
        }
    }
}

// ---------------------------------------------------------------------------
extern "C" void kernel_launch(void* const* d_in, const int* in_sizes, int n_in,
                              void* d_out, int out_size, void* d_ws, size_t ws_size,
                              hipStream_t stream) {
    (void)in_sizes; (void)n_in; (void)out_size; (void)ws_size;

    const float* pred_logits = (const float*)d_in[0];  // [BS,Q,NC]
    const float* pred_boxes  = (const float*)d_in[1];  // [BS,Q,4]
    const int*   tgt_labels  = (const int*)d_in[2];    // [SUMT]
    const float* tgt_bbox    = (const float*)d_in[3];  // [SUMT,4]

    float* Cout     = (float*)d_out;                          // [NROWS][SUMT]
    float* rows_out = Cout + (size_t)NROWS * SUMT;            // [BS][T]
    float* cols_out = rows_out + (size_t)BS * T;              // [BS][T]

    _Float16* probs = (_Float16*)d_ws;                        // [NROWS][NCP]

    // 1) softmax -> f16 probs (padded K)
    hm_softmax_kernel<<<(NROWS + 255) / 256, 256, 0, stream>>>(pred_logits, probs);

    // 2) fused WMMA cost matrix: grid (32 col-groups x 900 row tiles)
    hm_cost_kernel<<<dim3(32, Q, 1), 256, 0, stream>>>(
        probs, pred_boxes, tgt_labels, tgt_bbox, Cout);

    // 3) per-image Hungarian assignment
    hm_lsa_kernel<<<BS, 256, 0, stream>>>(Cout, rows_out, cols_out);
}
